// TopKSparseAutoencoder_19086834663627
// MI455X (gfx1250) — compile-verified, matrix operand order checked by reading, not run on hardware
//
#include <hip/hip_runtime.h>

typedef __attribute__((ext_vector_type(16))) __bf16 v16bf;
typedef __attribute__((ext_vector_type(8)))  __bf16 bf16x8;
typedef __attribute__((ext_vector_type(4)))  __bf16 bf16x4;
typedef __attribute__((ext_vector_type(8)))  float  v8f;
typedef __attribute__((ext_vector_type(4)))  float  f32x4;
typedef __attribute__((ext_vector_type(4)))  unsigned u32x4;

#define CAP 64   // max survivors kept per row (k=32; fp32 ties are ~impossible)

// ---------------- Kernel 1: xn = (x - pre_bias) / max(||.||, eps), stored bf16 ----------------
__global__ void normalize_rows_kernel(const float* __restrict__ x,
                                      const float* __restrict__ pre_bias,
                                      __bf16* __restrict__ xn,
                                      int D) {
  const int row = blockIdx.x;
  const float* xr = x + (size_t)row * D;
  __shared__ float red[256];
  float ss = 0.f;
  for (int c = threadIdx.x; c < D; c += blockDim.x) {
    float v = xr[c] - pre_bias[c];
    ss += v * v;
  }
  red[threadIdx.x] = ss;
  __syncthreads();
  for (int s = 128; s > 0; s >>= 1) {
    if ((int)threadIdx.x < s) red[threadIdx.x] += red[threadIdx.x + s];
    __syncthreads();
  }
  const float inv = 1.f / fmaxf(sqrtf(red[0]), 1e-12f);
  __bf16* xo = xn + (size_t)row * D;
  for (int c = threadIdx.x; c < D; c += blockDim.x) {
    xo[c] = (__bf16)((xr[c] - pre_bias[c]) * inv);
  }
}

// ---------------- Kernel 2: enc_w fp32 -> bf16 (vectorized) ----------------
__global__ void f32_to_bf16_kernel(const float* __restrict__ in,
                                   __bf16* __restrict__ out, size_t n4) {
  size_t i = (size_t)blockIdx.x * blockDim.x + threadIdx.x;
  const size_t stride = (size_t)gridDim.x * blockDim.x;
  for (; i < n4; i += stride) {
    f32x4 v = *(const f32x4*)(in + i * 4);
    bf16x4 o;
    o[0] = (__bf16)v[0]; o[1] = (__bf16)v[1]; o[2] = (__bf16)v[2]; o[3] = (__bf16)v[3];
    *(bf16x4*)(out + i * 4) = o;
  }
}

// ---------------- Kernel 3: latents = xn @ enc_w^T + enc_b (bf16 WMMA, fp32 acc) ----------
// Block: 256 threads = 8 waves arranged 2(M) x 4(N). Block tile 64x256; wave tile 32x64.
// Per K-step: 2 A-frags + 4 B-frags (12 x b128) feed 8 WMMAs -> 1.5 loads/WMMA.
__global__ void __launch_bounds__(256)
encoder_gemm_kernel(const __bf16* __restrict__ A,   // N x D (xn, bf16)
                    const __bf16* __restrict__ W,   // L x D (enc_w, bf16)
                    const float*  __restrict__ bias,// L
                    float* __restrict__ out,        // N x L (latents)
                    int D, int L) {
  const int lane = threadIdx.x & 31;
  const int wave = threadIdx.x >> 5;
  const int m0 = blockIdx.x * 64  + (wave >> 2) * 32;
  const int n0 = blockIdx.y * 256 + (wave & 3) * 64;
  const int half = lane >> 4;   // 0: K 0-7,16-23 ; 1: K 8-15,24-31
  const int l15  = lane & 15;

  v8f acc00 = {}, acc01 = {}, acc02 = {}, acc03 = {};
  v8f acc10 = {}, acc11 = {}, acc12 = {}, acc13 = {};

  const __bf16* arow0 = A + (size_t)(m0 + l15) * D + half * 8;       // M rows 0..15
  const __bf16* arow1 = arow0 + (size_t)16 * D;                      // M rows 16..31
  const __bf16* wrow0 = W + (size_t)(n0 + 0 * 16 + l15) * D + half * 8;
  const __bf16* wrow1 = W + (size_t)(n0 + 1 * 16 + l15) * D + half * 8;
  const __bf16* wrow2 = W + (size_t)(n0 + 2 * 16 + l15) * D + half * 8;
  const __bf16* wrow3 = W + (size_t)(n0 + 3 * 16 + l15) * D + half * 8;

  union Frag { v16bf v; bf16x8 h[2]; };

#define LOADFRAG(dst, ptr)                              \
  { (dst).h[0] = *(const bf16x8*)((ptr) + k0);          \
    (dst).h[1] = *(const bf16x8*)((ptr) + k0 + 16); }

  for (int k0 = 0; k0 < D; k0 += 32) {
    Frag a0, a1, b0, b1, b2, b3;
    LOADFRAG(a0, arow0) LOADFRAG(a1, arow1)
    LOADFRAG(b0, wrow0) LOADFRAG(b1, wrow1)
    LOADFRAG(b2, wrow2) LOADFRAG(b3, wrow3)

    acc00 = __builtin_amdgcn_wmma_f32_16x16x32_bf16(false, a0.v, false, b0.v, (short)0, acc00, false, false);
    acc01 = __builtin_amdgcn_wmma_f32_16x16x32_bf16(false, a0.v, false, b1.v, (short)0, acc01, false, false);
    acc02 = __builtin_amdgcn_wmma_f32_16x16x32_bf16(false, a0.v, false, b2.v, (short)0, acc02, false, false);
    acc03 = __builtin_amdgcn_wmma_f32_16x16x32_bf16(false, a0.v, false, b3.v, (short)0, acc03, false, false);
    acc10 = __builtin_amdgcn_wmma_f32_16x16x32_bf16(false, a1.v, false, b0.v, (short)0, acc10, false, false);
    acc11 = __builtin_amdgcn_wmma_f32_16x16x32_bf16(false, a1.v, false, b1.v, (short)0, acc11, false, false);
    acc12 = __builtin_amdgcn_wmma_f32_16x16x32_bf16(false, a1.v, false, b2.v, (short)0, acc12, false, false);
    acc13 = __builtin_amdgcn_wmma_f32_16x16x32_bf16(false, a1.v, false, b3.v, (short)0, acc13, false, false);
  }
#undef LOADFRAG

  // C/D layout: VGPR r, lanes 0-15 -> M=r, lanes 16-31 -> M=8+r; N = lane&15.
#define STORE_TILE(accv, I, J)                                                  \
  { const int col = n0 + (J) * 16 + l15; const float bv = bias[col];            \
    const int mb = m0 + (I) * 16 + half * 8;                                    \
    _Pragma("unroll")                                                           \
    for (int r = 0; r < 8; ++r) out[(size_t)(mb + r) * L + col] = (accv)[r] + bv; }

  STORE_TILE(acc00, 0, 0) STORE_TILE(acc01, 0, 1) STORE_TILE(acc02, 0, 2) STORE_TILE(acc03, 0, 3)
  STORE_TILE(acc10, 1, 0) STORE_TILE(acc11, 1, 1) STORE_TILE(acc12, 1, 2) STORE_TILE(acc13, 1, 3)
#undef STORE_TILE
}

// ---------------- Kernel 4: exact top-k threshold per row (radix select in LDS) + mask ----
__global__ void topk_mask_kernel(float* __restrict__ latents,  // N x L, masked in place
                                 const int* __restrict__ kptr,
                                 int* __restrict__ counts,
                                 int* __restrict__ sidx,
                                 float* __restrict__ sval,
                                 int L) {
  extern __shared__ unsigned smem_u[];
  unsigned* keys = smem_u;            // L entries (96 KB for L=24576)
  unsigned* hist = smem_u + L;        // 256 bins
  __shared__ unsigned sh_sel, sh_krem, sh_cnt;
  __shared__ int   s_idx[CAP];
  __shared__ float s_val[CAP];

  const int row = blockIdx.x;
  const int K = kptr[0];
  float* lrow = latents + (size_t)row * L;
  const int step4 = blockDim.x * 4;

  // Stage row as order-preserving uint keys (larger float <-> larger key). b128 in/out.
  for (int i = threadIdx.x * 4; i < L; i += step4) {
    f32x4 v = *(const f32x4*)(lrow + i);
    u32x4 kk;
    #pragma unroll
    for (int j = 0; j < 4; ++j) {
      unsigned b = __float_as_uint(v[j]);
      kk[j] = (b & 0x80000000u) ? ~b : (b | 0x80000000u);
    }
    *(u32x4*)(keys + i) = kk;
  }
  if (threadIdx.x == 0) sh_cnt = 0;
  __syncthreads();

  unsigned prefix = 0;
  int kth = K;
  for (int pass = 0; pass < 4; ++pass) {
    const int shift = 24 - 8 * pass;
    const unsigned maskHigh = (pass == 0) ? 0u : (0xFFFFFFFFu << (shift + 8));
    for (int b = threadIdx.x; b < 256; b += blockDim.x) hist[b] = 0;
    __syncthreads();
    for (int i = threadIdx.x * 4; i < L; i += step4) {
      u32x4 uu = *(const u32x4*)(keys + i);
      #pragma unroll
      for (int j = 0; j < 4; ++j) {
        const unsigned u = uu[j];
        if ((u & maskHigh) == prefix)
          atomicAdd(&hist[(u >> shift) & 255u], 1u);
      }
    }
    __syncthreads();
    if (threadIdx.x == 0) {
      unsigned cum = 0;
      unsigned b = 255;
      for (;; --b) {
        const unsigned c = hist[b];
        if (cum + c >= (unsigned)kth || b == 0) { sh_sel = b; sh_krem = (unsigned)kth - cum; break; }
        cum += c;
      }
    }
    __syncthreads();
    prefix |= (sh_sel << shift);
    kth = (int)sh_krem;
    __syncthreads();
  }
  const unsigned kthKey = prefix;  // exact key of k-th largest; >= keeps ties like reference

  for (int i = threadIdx.x * 4; i < L; i += step4) {
    u32x4 uu = *(const u32x4*)(keys + i);
    f32x4 vv;
    #pragma unroll
    for (int j = 0; j < 4; ++j) {
      const unsigned u = uu[j];
      float v = 0.f;
      if (u >= kthKey) {
        const unsigned b = (u & 0x80000000u) ? (u ^ 0x80000000u) : ~u;
        v = __uint_as_float(b);
        const unsigned slot = atomicAdd(&sh_cnt, 1u);
        if (slot < CAP) { s_idx[slot] = i + j; s_val[slot] = v; }
      }
      vv[j] = v;
    }
    *(f32x4*)(lrow + i) = vv;
  }
  __syncthreads();
  const int cnt = (int)(sh_cnt < (unsigned)CAP ? sh_cnt : (unsigned)CAP);
  if (threadIdx.x == 0) counts[row] = cnt;
  if ((int)threadIdx.x < cnt) {
    sidx[row * CAP + threadIdx.x] = s_idx[threadIdx.x];
    sval[row * CAP + threadIdx.x] = s_val[threadIdx.x];
  }
}

// ---------------- Kernel 5: out = sparse @ enc_w + pre_bias (fp32 gather over survivors) --
__global__ void decode_kernel(const int* __restrict__ counts,
                              const int* __restrict__ sidx,
                              const float* __restrict__ sval,
                              const float* __restrict__ enc_w,   // L x D fp32
                              const float* __restrict__ pre_bias,
                              float* __restrict__ out,           // N x D
                              int D) {
  const int row = blockIdx.x;
  __shared__ int   cidx[CAP];
  __shared__ float cval[CAP];
  __shared__ int   ccnt;
  if (threadIdx.x == 0) ccnt = counts[row];
  __syncthreads();
  const int cnt = ccnt;
  if ((int)threadIdx.x < cnt) {
    cidx[threadIdx.x] = sidx[row * CAP + threadIdx.x];
    cval[threadIdx.x] = sval[row * CAP + threadIdx.x];
  }
  __syncthreads();
  for (int c = threadIdx.x; c < D; c += blockDim.x) {
    float acc = pre_bias[c];
    for (int s = 0; s < cnt; ++s)
      acc += cval[s] * enc_w[(size_t)cidx[s] * D + c];
    out[(size_t)row * D + c] = acc;
  }
}

extern "C" void kernel_launch(void* const* d_in, const int* in_sizes, int n_in,
                              void* d_out, int out_size, void* d_ws, size_t ws_size,
                              hipStream_t stream) {
  const float* x        = (const float*)d_in[0];
  const float* enc_w    = (const float*)d_in[1];
  const float* enc_b    = (const float*)d_in[2];
  const float* pre_bias = (const float*)d_in[3];
  const int*   kptr     = (const int*)d_in[4];

  const int D = in_sizes[3];                 // 768
  const int N = in_sizes[0] / D;             // 8192
  const int L = in_sizes[2];                 // 24576

  float* out    = (float*)d_out;             // N*D (first tuple element)
  float* sparse = out + (size_t)N * D;       // N*L (second tuple element)

  char* ws = (char*)d_ws;
  size_t off = 0;
  __bf16* xn = (__bf16*)(ws + off);          off += (size_t)N * D * sizeof(__bf16);
  __bf16* wb = (__bf16*)(ws + off);          off += (size_t)L * D * sizeof(__bf16);
  int*   counts = (int*)(ws + off);          off += (size_t)N * sizeof(int);
  int*   sidx   = (int*)(ws + off);          off += (size_t)N * CAP * sizeof(int);
  float* sval   = (float*)(ws + off);        off += (size_t)N * CAP * sizeof(float);
  (void)ws_size; (void)n_in; (void)out_size;

  // 1) normalize rows of x -> bf16
  normalize_rows_kernel<<<N, 256, 0, stream>>>(x, pre_bias, xn, D);

  // 2) enc_w -> bf16 (vectorized, 4 elems/thread/iter)
  f32_to_bf16_kernel<<<4096, 256, 0, stream>>>(enc_w, wb, (size_t)L * D / 4);

  // 3) encoder GEMM (WMMA bf16, fp32 acc), bias fused, writes latents into sparse region
  dim3 g(N / 64, L / 256);
  encoder_gemm_kernel<<<g, 256, 0, stream>>>(xn, wb, enc_b, sparse, D, L);

  // 4) top-k mask in place + survivor compaction (96 KB + 1 KB dynamic LDS per block)
  const size_t shmem = (size_t)L * sizeof(unsigned) + 256 * sizeof(unsigned);
  (void)hipFuncSetAttribute((const void*)topk_mask_kernel,
                            hipFuncAttributeMaxDynamicSharedMemorySize, (int)shmem);
  topk_mask_kernel<<<N, 256, shmem, stream>>>(sparse, kptr, counts, sidx, sval, L);

  // 5) sparse decode in fp32
  decode_kernel<<<N, 256, 0, stream>>>(counts, sidx, sval, enc_w, pre_bias, out, D);
}
